// AttentionBlock_33002528703213
// MI455X (gfx1250) — matrix operand order repaired and checked by reading, not compiled, not run on hardware
//
#include <hip/hip_runtime.h>
#include <hip/hip_bf16.h>

typedef __attribute__((ext_vector_type(16))) _Float16 v16h;
typedef __attribute__((ext_vector_type(8)))  _Float16 v8h;
typedef __attribute__((ext_vector_type(8)))  float    v8f;
typedef __attribute__((ext_vector_type(4)))  unsigned int v4u;
typedef __attribute__((ext_vector_type(8)))  int v8i;
typedef __attribute__((ext_vector_type(4)))  int v4i;

#define B_  16
#define C_  256
#define N_  1024
#define NH_ 4
#define HD_ 64

__device__ inline v8f wmma_f16(v16h a, v16h b, v8f c) {
  // D = A(16x32 f16) x B(32x16 f16) + C(16x16 f32)
  return __builtin_amdgcn_wmma_f32_16x16x32_f16(false, a, false, b, (short)0, c, false, false);
}

union HV16 { v16h v; v8h h[2]; };

// A fragment (16x32 f16) from row-major matrix; base points at (row0, k0).
// Lanes 0-15: M=lane, K {0..7, 16..23}; lanes 16-31: M=lane-16, K {8..15, 24..31}.
__device__ inline v16h load_a_frag(const _Float16* base, int ld) {
  const int l = threadIdx.x & 31;
  const _Float16* p = base + (l & 15) * ld + ((l >> 4) << 3);
  HV16 r;
  r.h[0] = *(const v8h*)(p);
  r.h[1] = *(const v8h*)(p + 16);
  return r.v;
}

// B fragment (32x16 f16) where each column of B is contiguous in memory
// ("Bt" row-major, ld = column stride). Lane n<16: col n, K {0..7, 16..23}; lane n+16: col n, K {8..15, 24..31}.
__device__ inline v16h load_b_frag(const _Float16* base, int ld) {
  const int l = threadIdx.x & 31;
  const _Float16* p = base + (l & 15) * ld + ((l >> 4) << 3);
  HV16 r;
  r.h[0] = *(const v8h*)(p);
  r.h[1] = *(const v8h*)(p + 16);
  return r.v;
}

// Double-buffered 16(M) x 256(K) x 64(N) GEMM core: acc[4] 16x16 tiles.
// A row-major (ld=C_), B given as Bt row-major (columns contiguous, ld=C_).
__device__ inline void gemm_k256_4n(const _Float16* __restrict__ a_base,
                                    const _Float16* __restrict__ bt_base,
                                    v8f acc[4]) {
  v16h a0 = load_a_frag(a_base + 0, C_);
  v16h b0[4];
#pragma unroll
  for (int t = 0; t < 4; ++t) b0[t] = load_b_frag(bt_base + (size_t)(t * 16) * C_, C_);
  for (int kk = 0; kk < C_; kk += 32) {
    const int nk = kk + 32;
    v16h a1 = a0;
    v16h b1[4];
#pragma unroll
    for (int t = 0; t < 4; ++t) b1[t] = b0[t];
    if (nk < C_) {                       // prefetch next K-step while WMMAs issue
      a1 = load_a_frag(a_base + nk, C_);
#pragma unroll
      for (int t = 0; t < 4; ++t) b1[t] = load_b_frag(bt_base + (size_t)(t * 16) * C_ + nk, C_);
      if (nk + 32 < C_) __builtin_prefetch(bt_base + nk + 32, 0, 0);  // global_prefetch_b8
    }
#pragma unroll
    for (int t = 0; t < 4; ++t) acc[t] = wmma_f16(a0, b0[t], acc[t]);
    a0 = a1;
#pragma unroll
    for (int t = 0; t < 4; ++t) b0[t] = b1[t];
  }
}

// ---------------- weight f32 -> f16 conversion ----------------
__global__ void k_convw(const float* __restrict__ wq, const float* __restrict__ wp,
                        _Float16* __restrict__ wq16, _Float16* __restrict__ wp16) {
  int i = blockIdx.x * 256 + threadIdx.x;
  if (i < 768 * 256) wq16[i] = (_Float16)wq[i];
  if (i < 256 * 256) wp16[i] = (_Float16)wp[i];
}

// ---------------- GroupNorm: x[b][c][n] -> xnT[b][n][c] (f16) ----------------
__global__ __launch_bounds__(256) void k_gnorm(const float* __restrict__ x,
                                               const float* __restrict__ gw,
                                               const float* __restrict__ gb,
                                               _Float16* __restrict__ xnT) {
  const int b = blockIdx.x >> 5;
  const int g = blockIdx.x & 31;
  const float* xb = x + ((size_t)b * C_ + g * 8) * N_;
  float s = 0.f, ss = 0.f;
  for (int i = threadIdx.x; i < 8 * N_; i += 256) { float v = xb[i]; s += v; ss += v * v; }
  __shared__ float r1[256], r2[256];
  r1[threadIdx.x] = s; r2[threadIdx.x] = ss;
  __syncthreads();
  for (int st = 128; st > 0; st >>= 1) {
    if (threadIdx.x < st) { r1[threadIdx.x] += r1[threadIdx.x + st]; r2[threadIdx.x] += r2[threadIdx.x + st]; }
    __syncthreads();
  }
  const float mean = r1[0] * (1.f / 8192.f);
  const float var  = r2[0] * (1.f / 8192.f) - mean * mean;
  const float rstd = rsqrtf(var + 1e-5f);
  for (int i = threadIdx.x; i < 8 * N_; i += 256) {
    const int c = g * 8 + (i >> 10), n = i & 1023;
    const float y = (xb[i] - mean) * rstd * gw[c] + gb[c];
    xnT[((size_t)b * N_ + n) * C_ + c] = (_Float16)y;
  }
}

// ---------------- QKV GEMM ----------------
// grid: 16 b x 48 o-tiles x 2 n-halves; wave covers 16(M) x 64(N).
__global__ __launch_bounds__(256) void k_qkv(const _Float16* __restrict__ wq,
                                             const _Float16* __restrict__ xnT,
                                             const float* __restrict__ qkv_b,
                                             _Float16* __restrict__ qT,
                                             _Float16* __restrict__ kT,
                                             _Float16* __restrict__ vmat) {
  const int b  = blockIdx.x / 96;
  const int r_ = blockIdx.x % 96;
  const int o0 = (r_ >> 1) * 16;
  const int wid = threadIdx.x >> 5;
  const int lane = threadIdx.x & 31;
  const int half = lane >> 4, nn = lane & 15;
  const int n_base = (r_ & 1) * 512 + wid * 64;
  v8f acc[4];
#pragma unroll
  for (int t = 0; t < 4; ++t) acc[t] = (v8f){0, 0, 0, 0, 0, 0, 0, 0};
  gemm_k256_4n(wq + (size_t)o0 * C_, xnT + ((size_t)b * N_ + n_base) * C_, acc);
#pragma unroll
  for (int t = 0; t < 4; ++t) {
#pragma unroll
    for (int r = 0; r < 8; ++r) {
      const int o = o0 + r + half * 8;
      const int n = n_base + t * 16 + nn;
      float val = acc[t][r] + qkv_b[o];
      if (o < 256) {                       // q (fold in 1/sqrt(hd))
        val *= 0.125f;
        const int h = o >> 6, d = o & 63;
        qT[(((size_t)b * NH_ + h) * N_ + n) * HD_ + d] = (_Float16)val;
      } else if (o < 512) {                // k
        const int o2 = o - 256; const int h = o2 >> 6, d = o2 & 63;
        kT[(((size_t)b * NH_ + h) * N_ + n) * HD_ + d] = (_Float16)val;
      } else {                             // v (d-major for PV B-operand)
        const int o2 = o - 512; const int h = o2 >> 6, d = o2 & 63;
        vmat[(((size_t)b * NH_ + h) * HD_ + d) * N_ + n] = (_Float16)val;
      }
    }
  }
}

// ---------------- TDM descriptor build + issue (2D tile, f16 data) ----------------
// Bit packing per CDNA5 ISA ch.8 (D# groups 0/1); pad fields produce the LDS row
// padding in hardware. Issued by one wave; completion via TENSORcnt.
__device__ inline void tdm_load_2d(const void* gaddr, unsigned lds_off,
                                   unsigned tile_d0, unsigned tile_d1,
                                   unsigned tensor_d0, unsigned tensor_d1,
                                   unsigned d0_stride,
                                   unsigned pad_interval_code, unsigned pad_amount_code) {
#if defined(__has_builtin) && __has_builtin(__builtin_amdgcn_tensor_load_to_lds)
  const unsigned long long ga = (unsigned long long)(uintptr_t)gaddr;
  v4u g0;
  g0[0] = 1u;                                                   // count=1, user descriptor
  g0[1] = lds_off;                                              // lds_addr (bytes)
  g0[2] = (unsigned)(ga & 0xffffffffu);                         // global_addr[31:0]
  g0[3] = (unsigned)((ga >> 32) & 0x01ffffffu) | (2u << 30);    // global_addr[56:32] | type=2
  unsigned w0 = (1u << 16);                                     // data_size=1 (2 bytes)
  if (pad_interval_code | pad_amount_code)
    w0 |= (1u << 20) | (pad_interval_code << 22) | (pad_amount_code << 25);  // pad_enable
  v8i g1;
  g1[0] = (int)w0;                                              // wg_mask=0 (not in cluster)
  g1[1] = (int)((tensor_d0 & 0xffffu) << 16);                   // tensor_dim0[15:0]
  g1[2] = (int)(((tensor_d0 >> 16) & 0xffffu) | ((tensor_d1 & 0xffffu) << 16));
  g1[3] = (int)(((tensor_d1 >> 16) & 0xffffu) | ((tile_d0 & 0xffffu) << 16));
  g1[4] = (int)(tile_d1 & 0xffffu);                             // tile_dim1; tile_dim2=0
  g1[5] = (int)d0_stride;                                       // tensor_dim0_stride[31:0]
  g1[6] = 0;                                                    // stride hi / dim1_stride lo
  g1[7] = 0;
  const v4i gz4 = {0, 0, 0, 0};
  const v8i gz8 = {0, 0, 0, 0, 0, 0, 0, 0};
  // 6-arg toolchain variant: (g0, g1, g2, g3, g4, cpol)
  __builtin_amdgcn_tensor_load_to_lds(g0, g1, gz4, gz4, gz8, 0);
#endif
}

// ---------------- Flash attention: K,V for one (b,h) resident in LDS via TDM ----------------
__global__ __launch_bounds__(256) void k_attn(const _Float16* __restrict__ qT,
                                              const _Float16* __restrict__ kT,
                                              const _Float16* __restrict__ vmat,
                                              _Float16* __restrict__ oT) {
  extern __shared__ _Float16 smem[];
  const int KLD = 72, VLD = 1040, PLD = 40;       // padded strides
  _Float16* ks = smem;                            // [1024][KLD]  (pad 8h per 64h row)
  _Float16* vs = ks + 1024 * KLD;                 // [64][VLD]    (pad 8h per 512h)
  _Float16* ps = vs + 64 * VLD;                   // [8 waves][16][PLD]

  const int b   = blockIdx.x >> 5;
  const int h   = (blockIdx.x >> 3) & 3;
  const int blk = blockIdx.x & 7;
  const size_t bh = (size_t)b * NH_ + h;
  const _Float16* kg = kT   + bh * N_ * HD_;
  const _Float16* vg = vmat + bh * HD_ * N_;

#if defined(__has_builtin) && __has_builtin(__builtin_amdgcn_tensor_load_to_lds)
  if (threadIdx.x < 32) {   // one wave drives the TDM
    // K^T: tile 64(d) x 1024(j); pad 4 DWORDs every 32 DWORDs -> 72-half rows.
    tdm_load_2d(kg, 0u,           HD_, N_, HD_, N_, HD_, /*intv 32dw*/4, /*pad 4dw*/3);
    // V:   tile 1024(j) x 64(d); pad 4 DWORDs every 256 DWORDs -> 1040-half rows.
    tdm_load_2d(vg, 1024u * KLD * 2u, N_, HD_, N_, HD_, N_, /*intv 256dw*/7, /*pad 4dw*/3);
    __builtin_amdgcn_s_wait_tensorcnt(0);
  }
#else
  for (int i = threadIdx.x; i < (N_ * HD_) / 8; i += 256) {
    v8h c = *(const v8h*)(kg + (size_t)i * 8);
    *(v8h*)(ks + (i >> 3) * KLD + (i & 7) * 8) = c;
  }
  for (int i = threadIdx.x; i < (HD_ * N_) / 8; i += 256) {
    v8h c = *(const v8h*)(vg + (size_t)i * 8);
    const int row = i >> 7, col = (i & 127) * 8;
    *(v8h*)(vs + row * VLD + col + (col >= 512 ? 8 : 0)) = c;
  }
#endif
  __syncthreads();

  const int wid  = threadIdx.x >> 5;
  const int lane = threadIdx.x & 31;
  const int half = lane >> 4, nn = lane & 15;
  const int i0 = blk * 128 + wid * 16;
  const _Float16* qg = qT + bh * N_ * HD_ + (size_t)i0 * HD_;
  const v16h aq0 = load_a_frag(qg + 0, HD_);
  const v16h aq1 = load_a_frag(qg + 32, HD_);

  float mrow[8], lrow[8];
  v8f oacc[4];
#pragma unroll
  for (int r = 0; r < 8; ++r) { mrow[r] = -3.0e38f; lrow[r] = 0.f; }
#pragma unroll
  for (int t = 0; t < 4; ++t) oacc[t] = (v8f){0, 0, 0, 0, 0, 0, 0, 0};

  _Float16* pw = ps + wid * 16 * PLD;

  for (int j0 = 0; j0 < N_; j0 += 32) {
    const int jpad = (j0 >= 512) ? 8 : 0;       // skip V mid-row pad
    v8f s0 = (v8f){0, 0, 0, 0, 0, 0, 0, 0};
    v8f s1 = s0;
    s0 = wmma_f16(aq0, load_b_frag(ks + (j0     ) * KLD + 0,  KLD), s0);
    s0 = wmma_f16(aq1, load_b_frag(ks + (j0     ) * KLD + 32, KLD), s0);
    s1 = wmma_f16(aq0, load_b_frag(ks + (j0 + 16) * KLD + 0,  KLD), s1);
    s1 = wmma_f16(aq1, load_b_frag(ks + (j0 + 16) * KLD + 32, KLD), s1);

    float fac[8];
#pragma unroll
    for (int r = 0; r < 8; ++r) {
      float mx = fmaxf(s0[r], s1[r]);
#pragma unroll
      for (int off = 8; off > 0; off >>= 1) mx = fmaxf(mx, __shfl_xor(mx, off, 16));
      const float mnew = fmaxf(mrow[r], mx);
      const float p0 = __expf(s0[r] - mnew);
      const float p1 = __expf(s1[r] - mnew);
      float psum = p0 + p1;
#pragma unroll
      for (int off = 8; off > 0; off >>= 1) psum += __shfl_xor(psum, off, 16);
      const float f = __expf(mrow[r] - mnew);
      lrow[r] = lrow[r] * f + psum;
      mrow[r] = mnew;
      fac[r] = f;
      pw[(r + half * 8) * PLD + nn]      = (_Float16)p0;
      pw[(r + half * 8) * PLD + 16 + nn] = (_Float16)p1;
    }
    asm volatile("s_wait_dscnt 0x0" ::: "memory");   // same-wave LDS store->load ordering
    const v16h pa = load_a_frag(pw, PLD);            // P as A-operand (16x32)
#pragma unroll
    for (int t = 0; t < 4; ++t) {
#pragma unroll
      for (int r = 0; r < 8; ++r) oacc[t][r] *= fac[r];
      oacc[t] = wmma_f16(pa, load_b_frag(vs + (t * 16) * VLD + j0 + jpad, VLD), oacc[t]);
    }
  }

#pragma unroll
  for (int t = 0; t < 4; ++t) {
#pragma unroll
    for (int r = 0; r < 8; ++r) {
      const int i = i0 + r + half * 8;
      const int c = h * HD_ + t * 16 + nn;
      oT[((size_t)b * N_ + i) * C_ + c] = (_Float16)(oacc[t][r] / lrow[r]);
    }
  }
}

// ---------------- proj GEMM + bias + residual ----------------
__global__ __launch_bounds__(256) void k_proj(const _Float16* __restrict__ wp,
                                              const _Float16* __restrict__ oT,
                                              const float* __restrict__ proj_b,
                                              const float* __restrict__ x,
                                              float* __restrict__ out) {
  const int b  = blockIdx.x / 32;
  const int r_ = blockIdx.x % 32;
  const int o0 = (r_ >> 1) * 16;
  const int wid = threadIdx.x >> 5;
  const int lane = threadIdx.x & 31;
  const int half = lane >> 4, nn = lane & 15;
  const int n_base = (r_ & 1) * 512 + wid * 64;
  v8f acc[4];
#pragma unroll
  for (int t = 0; t < 4; ++t) acc[t] = (v8f){0, 0, 0, 0, 0, 0, 0, 0};
  gemm_k256_4n(wp + (size_t)o0 * C_, oT + ((size_t)b * N_ + n_base) * C_, acc);
#pragma unroll
  for (int t = 0; t < 4; ++t) {
#pragma unroll
    for (int r = 0; r < 8; ++r) {
      const int o = o0 + r + half * 8;
      const int n = n_base + t * 16 + nn;
      const size_t idx = ((size_t)b * C_ + o) * N_ + n;
      out[idx] = x[idx] + acc[t][r] + proj_b[o];
    }
  }
}

extern "C" void kernel_launch(void* const* d_in, const int* in_sizes, int n_in,
                              void* d_out, int out_size, void* d_ws, size_t ws_size,
                              hipStream_t stream) {
  const float* x      = (const float*)d_in[0];
  const float* gn_w   = (const float*)d_in[1];
  const float* gn_b   = (const float*)d_in[2];
  const float* qkv_w  = (const float*)d_in[3];
  const float* qkv_b  = (const float*)d_in[4];
  const float* proj_w = (const float*)d_in[5];
  const float* proj_b = (const float*)d_in[6];
  float* out = (float*)d_out;

  char* w = (char*)d_ws;                         // ~40.5 MB workspace used
  _Float16* xnT  = (_Float16*)(w);               // [16][1024][256]   8 MB
  _Float16* qT   = (_Float16*)(w + 8388608);     // [16][4][1024][64] 8 MB
  _Float16* kT   = (_Float16*)(w + 16777216);    // [16][4][1024][64] 8 MB
  _Float16* vmat = (_Float16*)(w + 25165824);    // [16][4][64][1024] 8 MB
  _Float16* oT   = (_Float16*)(w + 33554432);    // [16][1024][256]   8 MB
  _Float16* wq16 = (_Float16*)(w + 41943040);    // [768][256]
  _Float16* wp16 = (_Float16*)(w + 42336256);    // [256][256]

  k_convw<<<768, 256, 0, stream>>>(qkv_w, proj_w, wq16, wp16);
  k_gnorm<<<512, 256, 0, stream>>>(x, gn_w, gn_b, xnT);
  k_qkv  <<<1536, 256, 0, stream>>>(wq16, xnT, qkv_b, qT, kT, vmat);
  const size_t smem = (size_t)(1024 * 72 + 64 * 1040 + 8 * 16 * 40) * sizeof(_Float16); // 290,816 B
  k_attn <<<512, 256, smem, stream>>>(qT, kT, vmat, oT);
  k_proj <<<512, 256, 0, stream>>>(wp16, oT, proj_b, x, out);
}
